// IncrementalVGG16V2_9689446220305
// MI455X (gfx1250) — matrix operand order repaired and checked by reading, not compiled, not run on hardware
//
#include <hip/hip_runtime.h>
#include <hip/hip_bf16.h>

// ---------------------------------------------------------------------------
// CDNA5 (gfx1250) VGG16 forward: bf16 WMMA implicit-GEMM convs with weights
// double-buffered through LDS via GLOBAL_LOAD_ASYNC_TO_LDS_B128 (ASYNCcnt),
// 4-wave workgroups sharing the B slice; packed-B GEMM for layer 1; WMMA FCs.
// ---------------------------------------------------------------------------

typedef __attribute__((ext_vector_type(16))) __bf16    v16bf;
typedef __attribute__((ext_vector_type(8)))  float     v8f;
typedef __attribute__((ext_vector_type(4)))  unsigned  u32x4;   // trivial 16B vector

union Op16 {                 // one WMMA 16-bit operand: 16 halves / lane = 8 VGPRs
    v16bf          v;
    u32x4          q[2];
    unsigned short u[16];
};

__device__ __forceinline__ u32x4 ldq(const __hip_bfloat16* p) {
    return *reinterpret_cast<const u32x4*>(p);
}

__device__ __forceinline__ unsigned short f2bf(float f) {  // round-to-nearest-even
    union { float f; unsigned u; } x; x.f = f;
    unsigned r = x.u + 0x7FFFu + ((x.u >> 16) & 1u);
    return (unsigned short)(r >> 16);
}

// async 16-byte global -> LDS copy (GV mode, tracked by ASYNCcnt)
__device__ __forceinline__ void async_cp16(unsigned lds_off, const void* gsrc) {
    asm volatile("global_load_async_to_lds_b128 %0, %1, off"
                 :: "v"(lds_off), "v"((unsigned long long)gsrc) : "memory");
}
__device__ __forceinline__ void wait_async_le2() {
    asm volatile("s_wait_asynccnt 0x2" ::: "memory");
}
__device__ __forceinline__ void wait_async_0() {
    asm volatile("s_wait_asynccnt 0x0" ::: "memory");
}

// --- pack conv weights OIHW f32 -> WMMA-B lane order [chunk][coutTile][lane][16]
// K ordering: k = (ky*3+kx)*Cin + ci  (channel-minor so A chunks are contiguous)
__global__ void prep_conv_w(const float* __restrict__ w,
                            __hip_bfloat16* __restrict__ wpack,
                            int Cin, int Cout, int nChunks)
{
    int idx = blockIdx.x * blockDim.x + threadIdx.x;
    int nCt = Cout >> 4;
    int total = nChunks * nCt * 512;
    if (idx >= total) return;
    int i    = idx & 15;
    int lane = (idx >> 4) & 31;
    int ctc  = idx >> 9;            // chunk*nCt + ct
    int ct   = ctc % nCt;
    int c    = ctc / nCt;
    int hi = lane >> 4, ln = lane & 15;
    int cout = ct * 16 + ln;
    int k = c * 32 + hi * 8 + i + (i >= 8 ? 8 : 0);   // B layout: half i holds k = hi*8+i (i<8), 16+hi*8+(i-8) (i>=8) of chunk
    float val = 0.f;
    int K = 9 * Cin;
    if (k < K) {
        int r  = k / Cin, ci = k - r * Cin;
        int ky = r / 3,   kx = r - ky * 3;
        val = w[(((size_t)cout * Cin + ci) * 3 + ky) * 3 + kx];
    }
    wpack[idx] = __float2bfloat16(val);
}

// --- layer-1 im2col: f32 NCHW image -> bf16 A matrix [16*224*224][32]
__global__ void im2col_l1(const float* __restrict__ x,
                          __hip_bfloat16* __restrict__ out, int total)
{
    int idx = blockIdx.x * blockDim.x + threadIdx.x;
    if (idx >= total) return;
    int k = idx & 31;  int p = idx >> 5;
    int n = p / (224 * 224);
    int rem = p - n * 224 * 224;
    int y = rem / 224, xx = rem - (rem / 224) * 224;
    float v = 0.f;
    if (k < 27) {
        int r = k / 3, ci = k - r * 3;
        int iy = y + r / 3 - 1, ix = xx + (r - (r / 3) * 3) - 1;
        if (iy >= 0 && iy < 224 && ix >= 0 && ix < 224)
            v = x[(((size_t)n * 3 + ci) * 224 + iy) * 224 + ix];
    }
    out[idx] = __float2bfloat16(v);
}

// --- 2x2 stride-2 maxpool, NHWC bf16
__global__ void maxpool2(const __hip_bfloat16* __restrict__ in,
                         __hip_bfloat16* __restrict__ out,
                         int Ho, int Wo, int C, int total)
{
    int idx = blockIdx.x * blockDim.x + threadIdx.x;
    if (idx >= total) return;
    int c = idx % C;  int t = idx / C;
    int x = t % Wo;   t /= Wo;
    int y = t % Ho;   int n = t / Ho;
    int H = Ho * 2, W = Wo * 2;
    const __hip_bfloat16* p = in + ((size_t)(n * H + 2 * y) * W + 2 * x) * C + c;
    float a = __bfloat162float(p[0]);
    float b = __bfloat162float(p[C]);
    float d0 = __bfloat162float(p[(size_t)W * C]);
    float d1 = __bfloat162float(p[(size_t)W * C + C]);
    out[idx] = __float2bfloat16(fmaxf(fmaxf(a, b), fmaxf(d0, d1)));
}

// --- pool5 NHWC [16,7,7,512] -> reference (NCHW) flatten order: c*49+y*7+x
__global__ void flatten_to_nchw(const __hip_bfloat16* __restrict__ in,
                                __hip_bfloat16* __restrict__ out, int total)
{
    int idx = blockIdx.x * blockDim.x + threadIdx.x;
    if (idx >= total) return;
    int d = idx % 25088;  int n = idx / 25088;
    int c = d / 49;       int rem = d % 49;
    int y = rem / 7;      int x = rem % 7;
    out[idx] = in[((size_t)(n * 7 + y) * 7 + x) * 512 + c];
}

// --- plain GEMM with packed B (layer 1 after im2col, nChunks==1)
__global__ __launch_bounds__(32)
void gemm_nt4_wmma(const __hip_bfloat16* __restrict__ A,
                   const __hip_bfloat16* __restrict__ wpack,
                   const float* __restrict__ bias,
                   __hip_bfloat16* __restrict__ out,
                   int K, int N, int nChunks)
{
    const int lane = threadIdx.x;
    const int hi = lane >> 4, ln = lane & 15;
    const int tile = blockIdx.x;
    const int ct0 = blockIdx.y << 2;       // 4 cout tiles = 64 columns
    const int nCt = N >> 4;
    const __hip_bfloat16* arow = A + (size_t)(tile * 16 + ln) * K + (hi << 3);

    v8f acc0 = {}, acc1 = {}, acc2 = {}, acc3 = {};
    for (int c = 0; c < nChunks; ++c) {
        Op16 a;
        a.q[0] = ldq(arow + c * 32);
        a.q[1] = ldq(arow + c * 32 + 16);
        Op16 b[4];
        const __hip_bfloat16* wp =
            wpack + (((size_t)c * nCt + ct0) * 32 + lane) * 16;
#pragma unroll
        for (int j = 0; j < 4; ++j) {
            b[j].q[0] = ldq(wp + j * 512);
            b[j].q[1] = ldq(wp + j * 512 + 8);
        }
        acc0 = __builtin_amdgcn_wmma_f32_16x16x32_bf16(false, a.v, false, b[0].v, (short)0, acc0, false, false);
        acc1 = __builtin_amdgcn_wmma_f32_16x16x32_bf16(false, a.v, false, b[1].v, (short)0, acc1, false, false);
        acc2 = __builtin_amdgcn_wmma_f32_16x16x32_bf16(false, a.v, false, b[2].v, (short)0, acc2, false, false);
        acc3 = __builtin_amdgcn_wmma_f32_16x16x32_bf16(false, a.v, false, b[3].v, (short)0, acc3, false, false);
    }
    int cb = ct0 * 16 + ln;
    float bi0 = bias[cb], bi1 = bias[cb + 16], bi2 = bias[cb + 32], bi3 = bias[cb + 48];
#pragma unroll
    for (int rr = 0; rr < 8; ++rr) {
        size_t ob = (size_t)(tile * 16 + rr + (hi << 3)) * N + cb;
        float v0 = acc0[rr] + bi0; out[ob]      = __float2bfloat16(v0 > 0.f ? v0 : 0.f);
        float v1 = acc1[rr] + bi1; out[ob + 16] = __float2bfloat16(v1 > 0.f ? v1 : 0.f);
        float v2 = acc2[rr] + bi2; out[ob + 32] = __float2bfloat16(v2 > 0.f ? v2 : 0.f);
        float v3 = acc3[rr] + bi3; out[ob + 48] = __float2bfloat16(v3 > 0.f ? v3 : 0.f);
    }
}

// --- implicit-GEMM 3x3 SAME conv + bias + ReLU (Cin % 32 == 0 only).
// 4 waves/WG: wave w = pixel tile blockIdx.x*4+w, all share 64 couts.
// B slice (4 KB/chunk, contiguous in wpack) double-buffered in LDS via
// async-to-LDS copies; A = branchless masked global b128 loads per wave.
__global__ __launch_bounds__(128)
void conv3x3_wmma(const __hip_bfloat16* __restrict__ in,
                  const __hip_bfloat16* __restrict__ wpack,
                  const float* __restrict__ bias,
                  __hip_bfloat16* __restrict__ out,
                  int H, int W, int Cin, int Cout, int shR)
{
    __shared__ __hip_bfloat16 bstage[2][2048];      // 2 x 4KB B slices

    const int tid  = threadIdx.x;
    const int wv   = tid >> 5;
    const int lane = tid & 31;
    const int hi = lane >> 4, ln = lane & 15;
    const int tile = blockIdx.x * 4 + wv;          // this wave's 16-pixel tile
    const int ct0  = blockIdx.y << 2;              // 4 cout tiles
    const int nCt  = Cout >> 4;
    const int HW   = H * W;
    const int mCC  = (1 << shR) - 1;
    const int nChunks = 9 << shR;                  // even (shR >= 1)

    int p   = tile * 16 + ln;
    int n   = p / HW;
    int rem = p - n * HW;
    int y   = rem / W;
    int x   = rem - y * W;
    const size_t safeOff = ((size_t)(n * H + y) * W + x) * Cin;  // always valid

    v8f acc0 = {}, acc1 = {}, acc2 = {}, acc3 = {};
    Op16 a0, a1, b[4];

    auto stage = [&](int t, int buf) {             // 128 threads x 32B = 4KB
        const __hip_bfloat16* src = wpack + ((size_t)t * nCt + ct0) * 512 + tid * 16;
        unsigned dst = (unsigned)(size_t)&bstage[buf][tid * 16];
        async_cp16(dst,      src);
        async_cp16(dst + 16, src + 8);
    };
    auto loadA = [&](int t, Op16& a) {
        int r   = t >> shR;                        // filter tap 0..8 (uniform)
        int ci0 = (t & mCC) << 5;                  // first channel of chunk
        int dy = r / 3 - 1, dx = r - (r / 3) * 3 - 1;
        int iy = y + dy, ix = x + dx;
        bool valid = (iy >= 0) & (iy < H) & (ix >= 0) & (ix < W);
        size_t off = valid ? ((size_t)(n * H + iy) * W + ix) * Cin : safeOff;
        const __hip_bfloat16* ap = in + off + ci0 + (hi << 3);
        unsigned mm = valid ? ~0u : 0u;            // branchless SAME padding
        u32x4 m4 = {mm, mm, mm, mm};
        a.q[0] = ldq(ap) & m4;
        a.q[1] = ldq(ap + 16) & m4;
    };
    auto readB = [&](int buf) {                    // ds_load_b128 x8 per wave
        const __hip_bfloat16* bp = &bstage[buf][lane * 16];
#pragma unroll
        for (int j = 0; j < 4; ++j) {
            b[j].q[0] = *reinterpret_cast<const u32x4*>(bp + j * 512);
            b[j].q[1] = *reinterpret_cast<const u32x4*>(bp + j * 512 + 8);
        }
    };
    auto mma4 = [&](Op16& a) {
        acc0 = __builtin_amdgcn_wmma_f32_16x16x32_bf16(false, a.v, false, b[0].v, (short)0, acc0, false, false);
        acc1 = __builtin_amdgcn_wmma_f32_16x16x32_bf16(false, a.v, false, b[1].v, (short)0, acc1, false, false);
        acc2 = __builtin_amdgcn_wmma_f32_16x16x32_bf16(false, a.v, false, b[2].v, (short)0, acc2, false, false);
        acc3 = __builtin_amdgcn_wmma_f32_16x16x32_bf16(false, a.v, false, b[3].v, (short)0, acc3, false, false);
    };

    stage(0, 0);
    loadA(0, a0);
    for (int t = 0; t < nChunks; t += 2) {
        stage(t + 1, 1);                           // prefetch next B slice
        loadA(t + 1, a1);
        wait_async_le2();                          // retire stage(t) only
        __syncthreads();                           // buf0 visible to all waves
        readB(0);
        mma4(a0);
        __syncthreads();                           // all waves done reading buf0
        if (t + 2 < nChunks) {
            stage(t + 2, 0);
            loadA(t + 2, a0);
            wait_async_le2();                      // retire stage(t+1)
        } else {
            wait_async_0();
        }
        __syncthreads();                           // buf1 visible
        readB(1);
        mma4(a1);
        __syncthreads();                           // all waves done reading buf1
    }

    // epilogue: bias + ReLU + store NHWC bf16.  D: VGPR r -> M=r+8*hi, N=lane&15
    int cb = ct0 * 16 + ln;
    float bi0 = bias[cb], bi1 = bias[cb + 16], bi2 = bias[cb + 32], bi3 = bias[cb + 48];
#pragma unroll
    for (int rr = 0; rr < 8; ++rr) {
        size_t ob = (size_t)(tile * 16 + rr + (hi << 3)) * Cout + cb;
        float v0 = acc0[rr] + bi0; out[ob]      = __float2bfloat16(v0 > 0.f ? v0 : 0.f);
        float v1 = acc1[rr] + bi1; out[ob + 16] = __float2bfloat16(v1 > 0.f ? v1 : 0.f);
        float v2 = acc2[rr] + bi2; out[ob + 32] = __float2bfloat16(v2 > 0.f ? v2 : 0.f);
        float v3 = acc3[rr] + bi3; out[ob + 48] = __float2bfloat16(v3 > 0.f ? v3 : 0.f);
    }
}

// --- FC GEMM: M=16 (batch, single WMMA tile), N=Dout, K=Din (memory-bound).
__global__ __launch_bounds__(32)
void fc_wmma(const __hip_bfloat16* __restrict__ A,
             const float* __restrict__ Wt,
             const float* __restrict__ bias,
             __hip_bfloat16* __restrict__ outBf,
             float* __restrict__ outF,
             int K, int Dout, int doRelu)
{
    const int lane = threadIdx.x;
    const int hi = lane >> 4, ln = lane & 15;
    const int col0 = blockIdx.x * 32 + ln;
    const int col1 = col0 + 16;
    v8f acc0 = {}, acc1 = {};
    const int nChunks = K >> 5;

    for (int c = 0; c < nChunks; ++c) {
        Op16 a;
        const __hip_bfloat16* ap = A + (size_t)ln * K + c * 32 + (hi << 3);
        a.q[0] = ldq(ap);
        a.q[1] = ldq(ap + 16);

        Op16 b0, b1;
#pragma unroll
        for (int i = 0; i < 16; ++i) {
            int k = c * 32 + hi * 8 + i + (i >= 8 ? 8 : 0);
            const float* wr = Wt + (size_t)k * Dout;
            b0.u[i] = (col0 < Dout) ? f2bf(wr[col0]) : (unsigned short)0;
            b1.u[i] = (col1 < Dout) ? f2bf(wr[col1]) : (unsigned short)0;
        }
        acc0 = __builtin_amdgcn_wmma_f32_16x16x32_bf16(false, a.v, false, b0.v,
                                                       (short)0, acc0, false, false);
        acc1 = __builtin_amdgcn_wmma_f32_16x16x32_bf16(false, a.v, false, b1.v,
                                                       (short)0, acc1, false, false);
    }

    float bia0 = (col0 < Dout) ? bias[col0] : 0.f;
    float bia1 = (col1 < Dout) ? bias[col1] : 0.f;
#pragma unroll
    for (int rr = 0; rr < 8; ++rr) {
        int m = rr + hi * 8;                    // batch row
        float v0 = acc0[rr] + bia0;
        float v1 = acc1[rr] + bia1;
        if (doRelu) { v0 = v0 > 0.f ? v0 : 0.f; v1 = v1 > 0.f ? v1 : 0.f; }
        if (outF) {
            if (col0 < Dout) outF[(size_t)m * Dout + col0] = v0;
            if (col1 < Dout) outF[(size_t)m * Dout + col1] = v1;
        } else {
            if (col0 < Dout) outBf[(size_t)m * Dout + col0] = __float2bfloat16(v0);
            if (col1 < Dout) outBf[(size_t)m * Dout + col1] = __float2bfloat16(v1);
        }
    }
}

// ---------------------------------------------------------------------------
extern "C" void kernel_launch(void* const* d_in, const int* in_sizes, int n_in,
                              void* d_out, int out_size, void* d_ws, size_t ws_size,
                              hipStream_t stream)
{
    (void)in_sizes; (void)n_in; (void)out_size; (void)ws_size;
    const float* x = (const float*)d_in[0];
    // d_in: 0 x, 1 locations, 2 p_height, 3 p_width,
    //       4..29 conv (w,b) x13, 30..35 fc (w,b) x3

    char* ws = (char*)d_ws;
    size_t off = 0;
    auto alloc = [&](size_t bytes) -> char* {
        char* p = ws + off;
        off += (bytes + 255) & ~(size_t)255;
        return p;
    };

    const size_t P1 = (size_t)16 * 224 * 224;               // batch pixels, layer 1
    const size_t ACT_ELEMS = P1 * 64;                       // largest NHWC stage
    __hip_bfloat16* actA = (__hip_bfloat16*)alloc(ACT_ELEMS * 2);
    __hip_bfloat16* actB = (__hip_bfloat16*)alloc(ACT_ELEMS * 2);
    __hip_bfloat16* col1 = (__hip_bfloat16*)alloc(P1 * 32 * 2);   // layer-1 im2col
    __hip_bfloat16* fcb0 = (__hip_bfloat16*)alloc((size_t)16 * 25088 * 2);
    __hip_bfloat16* fcb1 = (__hip_bfloat16*)alloc((size_t)16 * 4096 * 2);
    __hip_bfloat16* fcb2 = (__hip_bfloat16*)alloc((size_t)16 * 4096 * 2);

    // ---- layer 1 (Cin=3): im2col + packed-B GEMM -------------------------
    {
        const float* wf = (const float*)d_in[4];
        const float* bf = (const float*)d_in[5];
        size_t pElems = (size_t)1 * (64 / 16) * 512;        // nChunks=1, nCt=4
        __hip_bfloat16* wp = (__hip_bfloat16*)alloc(pElems * 2);
        prep_conv_w<<<((int)pElems + 255) / 256, 256, 0, stream>>>(wf, wp, 3, 64, 1);
        int total = (int)(P1 * 32);
        im2col_l1<<<(total + 255) / 256, 256, 0, stream>>>(x, col1, total);
        dim3 grid((unsigned)(P1 / 16), 1);
        gemm_nt4_wmma<<<grid, 32, 0, stream>>>(col1, wp, bf, actA, 32, 64, 1);
    }

    // ---- remaining conv/pool stack ---------------------------------------
    static const int cfg[] = {64,0,128,128,0,256,256,256,0,
                              512,512,512,0,512,512,512,0};   // after layer 1
    const int nCfg = (int)(sizeof(cfg) / sizeof(cfg[0]));

    __hip_bfloat16* cur = actA;
    __hip_bfloat16* nxt = actB;
    int H = 224, W = 224, C = 64, convIdx = 1;

    for (int li = 0; li < nCfg; ++li) {
        if (cfg[li] == 0) {
            int Ho = H / 2, Wo = W / 2;
            int total = 16 * Ho * Wo * C;
            maxpool2<<<(total + 255) / 256, 256, 0, stream>>>(cur, nxt, Ho, Wo, C, total);
            H = Ho; W = Wo;
            __hip_bfloat16* t = cur; cur = nxt; nxt = t;
        } else {
            int Cin = C, Cout = cfg[li];
            const float* wf = (const float*)d_in[4 + 2 * convIdx];
            const float* bf = (const float*)d_in[5 + 2 * convIdx];
            int cpr = Cin / 32;                       // power of two
            int shR = 0; while ((1 << shR) < cpr) ++shR;
            int nChunks = 9 << shR;
            int nCt = Cout / 16;
            size_t pElems = (size_t)nChunks * nCt * 512;
            __hip_bfloat16* wp = (__hip_bfloat16*)alloc(pElems * 2);
            prep_conv_w<<<((int)pElems + 255) / 256, 256, 0, stream>>>(wf, wp, Cin, Cout, nChunks);
            dim3 grid(H * W / 4, Cout / 64);          // 4 pixel tiles per WG
            conv3x3_wmma<<<grid, 128, 0, stream>>>(cur, wp, bf, nxt, H, W, Cin, Cout, shR);
            C = Cout; ++convIdx;
            __hip_bfloat16* t = cur; cur = nxt; nxt = t;
        }
    }

    // ---- classifier -------------------------------------------------------
    {
        int total = 16 * 25088;
        flatten_to_nchw<<<(total + 255) / 256, 256, 0, stream>>>(cur, fcb0, total);
    }
    fc_wmma<<<4096 / 32, 32, 0, stream>>>(fcb0, (const float*)d_in[30],
                                          (const float*)d_in[31], fcb1, nullptr,
                                          25088, 4096, 1);
    fc_wmma<<<4096 / 32, 32, 0, stream>>>(fcb1, (const float*)d_in[32],
                                          (const float*)d_in[33], fcb2, nullptr,
                                          4096, 4096, 1);
    fc_wmma<<<(1000 + 31) / 32, 32, 0, stream>>>(fcb2, (const float*)d_in[34],
                                                 (const float*)d_in[35], nullptr,
                                                 (float*)d_out, 4096, 1000, 0);
}
